// CLUBCategorical_68753836474458
// MI455X (gfx1250) — matrix operand order repaired and checked by reading, not compiled, not run on hardware
//
#include <hip/hip_runtime.h>
#include <stdint.h>

// ---------------------------------------------------------------------------
// CDNA5 (gfx1250) CLUB-categorical pipeline.
// out[i] = logits[i, y[i]] - (1/N) * sum_c count[c] * logits[i, c]
// (log-softmax terms cancel exactly since sum_c count[c] == N)
// GEMMs via v_wmma_f32_16x16x32_bf16 with split-bf16 (hi+lo) fp32 emulation,
// tiles staged with double-buffered GLOBAL_LOAD_ASYNC_TO_LDS_B128 (ASYNCcnt).
// ---------------------------------------------------------------------------

#define N_SAMPLES 8192
#define X_DIM     512
#define Y_DIM     512
#define HIDDEN    1024

typedef __attribute__((ext_vector_type(16))) __bf16 v16bf;
typedef __attribute__((ext_vector_type(8)))  float  v8f;

#if __has_builtin(__builtin_amdgcn_s_wait_asynccnt)
#define WAIT_ASYNCCNT(n) __builtin_amdgcn_s_wait_asynccnt(n)
#else
#define WAIT_ASYNCCNT(n) asm volatile("s_wait_asynccnt %0" ::"i"(n) : "memory")
#endif

static __device__ __forceinline__ unsigned short f32_to_bf16_rne(float f) {
  union { float f; uint32_t u; } c; c.f = f;
  uint32_t u = c.u;
  uint32_t r = u + 0x7FFFu + ((u >> 16) & 1u);
  if ((u & 0x7F800000u) == 0x7F800000u) r = u;   // inf/nan: truncate
  return (unsigned short)(r >> 16);
}
static __device__ __forceinline__ float bf16_to_f32(unsigned short h) {
  union { uint32_t u; float f; } c; c.u = ((uint32_t)h) << 16;
  return c.f;
}

// Low 32 bits of a generic pointer to __shared__ == LDS byte offset.
static __device__ __forceinline__ uint32_t lds_off(const void* p) {
  return (uint32_t)(uintptr_t)p;
}

// Async DMA: LDS[dst] = MEM[src], 16 bytes per lane, tracked by ASYNCcnt.
static __device__ __forceinline__ void async_copy_b128(uint32_t lds_dst,
                                                       const void* gsrc) {
  asm volatile("global_load_async_to_lds_b128 %0, %1, off"
               :: "v"(lds_dst), "v"((uint64_t)(uintptr_t)gsrc)
               : "memory");
}

// ---------------------------------------------------------------------------
// Histogram of y_idx
// ---------------------------------------------------------------------------
__global__ void zero_counts_kernel(int* counts) {
  counts[threadIdx.x] = 0;
}

__global__ void hist_kernel(const int* __restrict__ y, int* __restrict__ counts, int n) {
  int i = blockIdx.x * blockDim.x + threadIdx.x;
  if (i < n) atomicAdd(&counts[y[i]], 1);
}

// ---------------------------------------------------------------------------
// fp32 -> (bf16 hi, bf16 lo) split, same layout
// ---------------------------------------------------------------------------
__global__ void split_bf16_kernel(const float* __restrict__ src,
                                  unsigned short* __restrict__ hi,
                                  unsigned short* __restrict__ lo, int n) {
  int i = blockIdx.x * blockDim.x + threadIdx.x;
  if (i < n) {
    float v = src[i];
    unsigned short h = f32_to_bf16_rne(v);
    hi[i] = h;
    lo[i] = f32_to_bf16_rne(v - bf16_to_f32(h));
  }
}

// fp32 W[K][Nn] -> transposed bf16 hi/lo [Nn][K]
__global__ void split_bf16_transpose_kernel(const float* __restrict__ src,
                                            unsigned short* __restrict__ hi,
                                            unsigned short* __restrict__ lo,
                                            int K, int Nn) {
  int i = blockIdx.x * blockDim.x + threadIdx.x;
  if (i < K * Nn) {
    int k = i / Nn, n = i % Nn;
    float v = src[i];
    unsigned short h = f32_to_bf16_rne(v);
    size_t t = (size_t)n * K + k;
    hi[t] = h;
    lo[t] = f32_to_bf16_rne(v - bf16_to_f32(h));
  }
}

// ---------------------------------------------------------------------------
// Split-bf16 WMMA GEMM:  C[M][Nn] = A[M][K] * B^T[Nn][K] + bias
//   A given as (Ahi,Alo) row-major [M][K] bf16
//   B given as (Bhi,Blo) row-major [Nn][K] bf16 (i.e. original W transposed)
//   RELU_SPLIT: epilogue relu + re-split to (outHi,outLo); else fp32 out.
// Block: 256 threads (8 waves), tile 128x128, each wave 32x64 (2x4 frags).
// Double-buffered async global->LDS staging overlapped with WMMA.
// ---------------------------------------------------------------------------
union FragU { uint4 q[2]; v16bf v; };

template<int K_TOTAL, bool RELU_SPLIT>
__global__ __launch_bounds__(256)
void gemm_wmma_kernel(const unsigned short* __restrict__ Ahi,
                      const unsigned short* __restrict__ Alo,
                      const unsigned short* __restrict__ Bhi,
                      const unsigned short* __restrict__ Blo,
                      const float* __restrict__ bias,
                      float* __restrict__ outF,
                      unsigned short* __restrict__ outHi,
                      unsigned short* __restrict__ outLo,
                      int Nn) {
  constexpr int LDT = 40;                       // padded LDS row stride (bf16)
  constexpr int TILE = 128 * LDT;               // ushorts per tile array
  __shared__ unsigned short sAh[2][TILE];
  __shared__ unsigned short sAl[2][TILE];
  __shared__ unsigned short sBh[2][TILE];
  __shared__ unsigned short sBl[2][TILE];

  const int tid  = threadIdx.x;
  const int lane = tid & 31;
  const int wave = tid >> 5;
  const int wm   = wave >> 1;                   // 0..3 -> M sub-tile
  const int wn   = wave & 1;                    // 0..1 -> N sub-tile
  const int lm   = lane & 15;
  const int half = lane >> 4;

  const int m0 = blockIdx.x * 128;
  const int n0 = blockIdx.y * 128;

  // Async-copy lane mapping: each instruction moves 8 rows x 32 bf16
  // (lane -> row = lane/4, 16B chunk = (lane&3)*8 elements).
  const int arow = lane >> 2;
  const int acol = (lane & 3) * 8;

  // Issue 8 async b128 copies (this wave's 16 rows of all 4 tile arrays).
  auto issue_tile = [&](int buf, int k0) {
    #pragma unroll
    for (int r8 = 0; r8 < 2; ++r8) {
      int row  = wave * 16 + r8 * 8 + arow;     // 0..127
      int dst  = row * LDT + acol;              // bf16 element offset in tile
      size_t ga = (size_t)(m0 + row) * K_TOTAL + k0 + acol;
      size_t gb = (size_t)(n0 + row) * K_TOTAL + k0 + acol;
      async_copy_b128(lds_off(&sAh[buf][dst]), Ahi + ga);
      async_copy_b128(lds_off(&sAl[buf][dst]), Alo + ga);
      async_copy_b128(lds_off(&sBh[buf][dst]), Bhi + gb);
      async_copy_b128(lds_off(&sBl[buf][dst]), Blo + gb);
    }
  };

  v8f acc[2][4];
  const v8f vzero = {0.f, 0.f, 0.f, 0.f, 0.f, 0.f, 0.f, 0.f};
#pragma unroll
  for (int i = 0; i < 2; ++i)
#pragma unroll
    for (int j = 0; j < 4; ++j) acc[i][j] = vzero;

  constexpr int NSTEP = K_TOTAL / 32;
  issue_tile(0, 0);                             // prologue: fill buffer 0

  for (int s = 0; s < NSTEP; ++s) {
    const int buf = s & 1;
    if (s + 1 < NSTEP) {
      issue_tile(buf ^ 1, (s + 1) * 32);        // prefetch next tile (DMA)
      WAIT_ASYNCCNT(8);                         // current buffer's 8 are done
    } else {
      WAIT_ASYNCCNT(0);
    }
    __syncthreads();                            // all waves' tiles visible

    const unsigned short* pAh = sAh[buf];
    const unsigned short* pAl = sAl[buf];
    const unsigned short* pBh = sBh[buf];
    const unsigned short* pBl = sBl[buf];

    // A fragment (16x32 bf16): lane holds row m, K chunks at half*8, 16+half*8
    FragU fAh[2], fAl[2];
#pragma unroll
    for (int mf = 0; mf < 2; ++mf) {
      int m = wm * 32 + mf * 16 + lm;
      int base = m * LDT + half * 8;
      fAh[mf].q[0] = *(const uint4*)(pAh + base);
      fAh[mf].q[1] = *(const uint4*)(pAh + base + 16);
      fAl[mf].q[0] = *(const uint4*)(pAl + base);
      fAl[mf].q[1] = *(const uint4*)(pAl + base + 16);
    }
    // B fragment (32x16 bf16): lane holds col n, K chunks at half*8, 16+half*8
    FragU fBh[4], fBl[4];
#pragma unroll
    for (int nf = 0; nf < 4; ++nf) {
      int n = wn * 64 + nf * 16 + lm;
      int base = n * LDT + half * 8;
      fBh[nf].q[0] = *(const uint4*)(pBh + base);
      fBh[nf].q[1] = *(const uint4*)(pBh + base + 16);
      fBl[nf].q[0] = *(const uint4*)(pBl + base);
      fBl[nf].q[1] = *(const uint4*)(pBl + base + 16);
    }

    // Split-bf16 product: Ahi*Bhi + Ahi*Blo + Alo*Bhi
#pragma unroll
    for (int mf = 0; mf < 2; ++mf)
#pragma unroll
      for (int nf = 0; nf < 4; ++nf) {
        acc[mf][nf] = __builtin_amdgcn_wmma_f32_16x16x32_bf16(
            false, fAh[mf].v, false, fBh[nf].v, (short)0, acc[mf][nf], false, false);
        acc[mf][nf] = __builtin_amdgcn_wmma_f32_16x16x32_bf16(
            false, fAh[mf].v, false, fBl[nf].v, (short)0, acc[mf][nf], false, false);
        acc[mf][nf] = __builtin_amdgcn_wmma_f32_16x16x32_bf16(
            false, fAl[mf].v, false, fBh[nf].v, (short)0, acc[mf][nf], false, false);
      }

    __syncthreads();   // all reads of `buf` done before step s+1 overwrites it
  }

  // Epilogue. C/D layout: lane -> n = lane%16; VGPR v -> m = v + 8*(lane/16)
#pragma unroll
  for (int mf = 0; mf < 2; ++mf)
#pragma unroll
    for (int nf = 0; nf < 4; ++nf) {
      int col = n0 + wn * 64 + nf * 16 + lm;
      float bv = bias[col];
#pragma unroll
      for (int v = 0; v < 8; ++v) {
        int row = m0 + wm * 32 + mf * 16 + v + half * 8;
        float val = acc[mf][nf][v] + bv;
        size_t idx = (size_t)row * Nn + col;
        if (RELU_SPLIT) {
          val = fmaxf(val, 0.0f);
          unsigned short h = f32_to_bf16_rne(val);
          outHi[idx] = h;
          outLo[idx] = f32_to_bf16_rne(val - bf16_to_f32(h));
        } else {
          outF[idx] = val;
        }
      }
    }
}

// ---------------------------------------------------------------------------
// out[i] = logits[i, y[i]] - (1/N) * sum_c count[c]*logits[i, c]
// one wave (32 lanes) per row
// ---------------------------------------------------------------------------
__global__ void loss_kernel(const float* __restrict__ logits,
                            const int* __restrict__ y,
                            const int* __restrict__ counts,
                            float* __restrict__ out) {
  int wave = threadIdx.x >> 5;
  int lane = threadIdx.x & 31;
  int row  = blockIdx.x * 8 + wave;
  if (row >= N_SAMPLES) return;
  const float* lr = logits + (size_t)row * Y_DIM;
  float s = 0.0f;
#pragma unroll
  for (int c = lane; c < Y_DIM; c += 32)
    s += (float)counts[c] * lr[c];
#pragma unroll
  for (int off = 16; off > 0; off >>= 1)
    s += __shfl_xor(s, off, 32);
  if (lane == 0)
    out[row] = lr[y[row]] - s * (1.0f / (float)N_SAMPLES);
}

// ---------------------------------------------------------------------------
extern "C" void kernel_launch(void* const* d_in, const int* in_sizes, int n_in,
                              void* d_out, int out_size, void* d_ws, size_t ws_size,
                              hipStream_t stream) {
  (void)in_sizes; (void)n_in; (void)out_size; (void)ws_size;

  const float* x  = (const float*)d_in[0];   // [8192, 512]
  const int*   y  = (const int*)  d_in[1];   // [8192]
  const float* W1 = (const float*)d_in[2];   // [512, 1024]
  const float* b1 = (const float*)d_in[3];   // [1024]
  const float* W2 = (const float*)d_in[4];   // [1024, 512]
  const float* b2 = (const float*)d_in[5];   // [512]
  float* out = (float*)d_out;                // [8192]

  // Workspace carve-up (256B aligned)
  char* base = (char*)d_ws;
  size_t off = 0;
  auto carve = [&](size_t bytes) -> char* {
    char* p = base + off;
    off = (off + bytes + 255) & ~(size_t)255;
    return p;
  };
  int*            counts = (int*)           carve(Y_DIM * sizeof(int));
  unsigned short* x_hi   = (unsigned short*)carve((size_t)N_SAMPLES * X_DIM * 2);
  unsigned short* x_lo   = (unsigned short*)carve((size_t)N_SAMPLES * X_DIM * 2);
  unsigned short* w1t_hi = (unsigned short*)carve((size_t)HIDDEN * X_DIM * 2);
  unsigned short* w1t_lo = (unsigned short*)carve((size_t)HIDDEN * X_DIM * 2);
  unsigned short* w2t_hi = (unsigned short*)carve((size_t)Y_DIM * HIDDEN * 2);
  unsigned short* w2t_lo = (unsigned short*)carve((size_t)Y_DIM * HIDDEN * 2);
  unsigned short* h_hi   = (unsigned short*)carve((size_t)N_SAMPLES * HIDDEN * 2);
  unsigned short* h_lo   = (unsigned short*)carve((size_t)N_SAMPLES * HIDDEN * 2);
  float*          logits = (float*)         carve((size_t)N_SAMPLES * Y_DIM * 4);

  // 1) histogram of labels
  zero_counts_kernel<<<1, Y_DIM, 0, stream>>>(counts);
  hist_kernel<<<N_SAMPLES / 256, 256, 0, stream>>>(y, counts, N_SAMPLES);

  // 2) precision-split inputs / weights (weights also transposed to [N][K])
  {
    int n = N_SAMPLES * X_DIM;
    split_bf16_kernel<<<n / 256, 256, 0, stream>>>(x, x_hi, x_lo, n);
  }
  {
    int n = X_DIM * HIDDEN;
    split_bf16_transpose_kernel<<<n / 256, 256, 0, stream>>>(W1, w1t_hi, w1t_lo, X_DIM, HIDDEN);
  }
  {
    int n = HIDDEN * Y_DIM;
    split_bf16_transpose_kernel<<<n / 256, 256, 0, stream>>>(W2, w2t_hi, w2t_lo, HIDDEN, Y_DIM);
  }

  // 3) h = relu(x@W1 + b1), stored as split bf16
  gemm_wmma_kernel<X_DIM, true><<<dim3(N_SAMPLES / 128, HIDDEN / 128), 256, 0, stream>>>(
      x_hi, x_lo, w1t_hi, w1t_lo, b1, nullptr, h_hi, h_lo, HIDDEN);

  // 4) logits = h@W2 + b2 (fp32)
  gemm_wmma_kernel<HIDDEN, false><<<dim3(N_SAMPLES / 128, Y_DIM / 128), 256, 0, stream>>>(
      h_hi, h_lo, w2t_hi, w2t_lo, b2, logits, nullptr, nullptr, Y_DIM);

  // 5) final reduction
  loss_kernel<<<N_SAMPLES / 8, 256, 0, stream>>>(logits, y, counts, out);
}
